// EnhancedSAModule_61546881352072
// MI455X (gfx1250) — compile-verified, matrix operand order checked by reading, not run on hardware
//
#include <hip/hip_runtime.h>
#include <hip/hip_bf16.h>

typedef __attribute__((ext_vector_type(16))) _Float16 v16h;
typedef __attribute__((ext_vector_type(8)))  float    v8f;

#define N_PTS   32768
#define M_SMP   4096
#define KNB     32
#define EPS_LN  1e-5f

// ---- WMMA helpers -----------------------------------------------------------
// A-matrix 16x32 f16 layout (CDNA5 ISA 7.12.2): lane L -> row m = L%16,
// half = L/16; element e (pair j=e/2, p=e%2) -> K = (j>=4?16:0)+8*half+2*(j&3)+p
__device__ __forceinline__ int kofs(int e, int half) {
  int j = e >> 1, p = e & 1;
  return ((j >= 4) ? 16 : 0) + half * 8 + 2 * (j & 3) + p;
}

__device__ __forceinline__ v8f wmma16(v16h a, v16h b, v8f c) {
  return __builtin_amdgcn_wmma_f32_16x16x32_f16(false, a, false, b, (short)0, c,
                                                false, false);
}

// Load an A fragment from a row-major f16 buffer; rowk = &buf[row][ktile*32]
// (each half-fragment is 16 contiguous bytes -> lowers to 2x b128 loads)
__device__ __forceinline__ v16h a_frag(const _Float16* rowk, int half) {
  v16h a;
#pragma unroll
  for (int e = 0; e < 16; ++e) a[e] = rowk[kofs(e, half)];
  return a;
}

// ---- weight pre-swizzle: f32 row-major [K,N] -> f16 B-fragment-major --------
__global__ void frag_convert_kernel(const float* __restrict__ W,
                                    _Float16* __restrict__ Wf,
                                    int Kreal, int N) {
  int lane = threadIdx.x;
  int NT = N / 16;
  int kt = blockIdx.x / NT, nt = blockIdx.x % NT;
  int n = nt * 16 + (lane & 15), half = lane >> 4;
  v16h v;
#pragma unroll
  for (int e = 0; e < 16; ++e) {
    int k = kt * 32 + kofs(e, half);
    v[e] = (k < Kreal) ? (_Float16)W[(size_t)k * N + n] : (_Float16)0.0f;
  }
  ((v16h*)Wf)[(size_t)blockIdx.x * 32 + lane] = v;
}

// ---- KNN: one wave per query, per-lane sorted top-32 in LDS + wave merge ----
__global__ void knn_kernel(const float* __restrict__ pos,
                           const int* __restrict__ idx,
                           int* __restrict__ col) {
  int m = blockIdx.x, lane = threadIdx.x;
  int q = idx[m];
  float qx = pos[3 * q], qy = pos[3 * q + 1], qz = pos[3 * q + 2];
  __shared__ float sd[KNB * 32];
  __shared__ int   si[KNB * 32];
  for (int s = 0; s < KNB; ++s) { sd[s * 32 + lane] = 3.0e38f; si[s * 32 + lane] = 0; }
  for (int i = lane; i < N_PTS; i += 32) {
    float dx = pos[3 * i] - qx, dy = pos[3 * i + 1] - qy, dz = pos[3 * i + 2] - qz;
    float d = dx * dx + dy * dy + dz * dz;
    if (d < sd[(KNB - 1) * 32 + lane]) {
      int s = KNB - 1;
      while (s > 0 && sd[(s - 1) * 32 + lane] > d) {
        sd[s * 32 + lane] = sd[(s - 1) * 32 + lane];
        si[s * 32 + lane] = si[(s - 1) * 32 + lane];
        --s;
      }
      sd[s * 32 + lane] = d;
      si[s * 32 + lane] = i;
    }
  }
  int ptr = 0;
  for (int r = 0; r < KNB; ++r) {
    float v = (ptr < KNB) ? sd[ptr * 32 + lane] : 3.0e38f;
    int  iv = (ptr < KNB) ? si[ptr * 32 + lane] : 0;
    float mn = v;
#pragma unroll
    for (int off = 16; off >= 1; off >>= 1) mn = fminf(mn, __shfl_xor(mn, off, 32));
    unsigned long long ball = __ballot(v == mn);
    int winner = __ffsll(ball) - 1;
    int widx = __shfl(iv, winner, 32);
    if (lane == 0) col[m * KNB + r] = widx;
    if (lane == winner) ptr++;
  }
}

// ---- PointNetConv: relu(feat@W1+b1)@W2+b2, max over 32 neighbors -----------
// W1 fragments staged to LDS with the CDNA5 async-copy path (ASYNCcnt).
__global__ void conv_kernel(const float* __restrict__ x,
                            const float* __restrict__ pos,
                            const int* __restrict__ idx,
                            const int* __restrict__ col,
                            const _Float16* __restrict__ W1f, const float* __restrict__ b1,
                            const _Float16* __restrict__ W2f, const float* __restrict__ b2,
                            _Float16* __restrict__ agg16) {
  int m = blockIdx.x;
  int tid = threadIdx.x, lane = tid & 31, w = tid >> 5;
  __shared__ _Float16 feat[32][96];   // 32 neighbors x (64 feat + 3 rel + pad)
  __shared__ _Float16 hbuf[32][128];
  __shared__ __align__(32) _Float16 w1s[24 * 512];  // 24 KB: swizzled W1
  __shared__ int scol[32];
  __shared__ float qp[3];
  __builtin_prefetch(W2f, 0, 3);
  // async-copy W1 fragments global -> LDS (16B per lane per issue)
  {
    unsigned lbase = (unsigned)(uintptr_t)&w1s[0];
    for (int t = tid; t < (24 * 512) / 8; t += 256) {
      unsigned loff = lbase + (unsigned)t * 16u;
      unsigned long long ga = (unsigned long long)(uintptr_t)(W1f + t * 8);
      asm volatile("global_load_async_to_lds_b128 %0, %1, off"
                   :: "v"(loff), "v"(ga) : "memory");
    }
  }
  if (tid < 32) scol[tid] = col[m * KNB + tid];
  if (tid < 3)  qp[tid] = pos[3 * idx[m] + tid];
  __syncthreads();
  for (int t = tid; t < 32 * 96; t += 256) {
    int j = t / 96, d = t % 96;
    int c = scol[j];
    float v = 0.f;
    if (d < 64)      v = x[(size_t)c * 64 + d];
    else if (d < 67) v = pos[3 * c + (d - 64)] - qp[d - 64];
    feat[j][d] = (_Float16)v;
  }
  asm volatile("s_wait_asynccnt 0x0" ::: "memory");
  __syncthreads();
  // GEMM1: [32,96] @ [96,128], relu -> hbuf.  8 waves, one col-tile each.
  {
    int ct = w;
    int n = ct * 16 + (lane & 15), half = lane >> 4;
#pragma unroll
    for (int rt = 0; rt < 2; ++rt) {
      v8f acc = {};
#pragma unroll
      for (int kt = 0; kt < 3; ++kt) {
        v16h a = a_frag(&feat[rt * 16 + (lane & 15)][kt * 32], half);
        v16h b = ((const v16h*)w1s)[(kt * 8 + ct) * 32 + lane];
        acc = wmma16(a, b, acc);
      }
      float bv = b1[n];
#pragma unroll
      for (int r = 0; r < 8; ++r) {
        float h = acc[r] + bv;
        hbuf[rt * 16 + r + 8 * half][n] = (_Float16)(h > 0.f ? h : 0.f);
      }
    }
  }
  __syncthreads();
  // GEMM2: [32,128] @ [128,256]; column-max over all 32 rows -> agg
#pragma unroll
  for (int cc = 0; cc < 2; ++cc) {
    int ct = w * 2 + cc;
    int n = ct * 16 + (lane & 15), half = lane >> 4;
    v8f acc0 = {}, acc1 = {};
#pragma unroll
    for (int kt = 0; kt < 4; ++kt) {
      v16h b  = ((const v16h*)W2f)[(kt * 16 + ct) * 32 + lane];
      v16h a0 = a_frag(&hbuf[(lane & 15)][kt * 32], half);
      v16h a1 = a_frag(&hbuf[16 + (lane & 15)][kt * 32], half);
      acc0 = wmma16(a0, b, acc0);
      acc1 = wmma16(a1, b, acc1);
    }
    float bv = b2[n];
    float vmax = -3.0e38f;
#pragma unroll
    for (int r = 0; r < 8; ++r) {
      vmax = fmaxf(vmax, acc0[r] + bv);
      vmax = fmaxf(vmax, acc1[r] + bv);
    }
    vmax = fmaxf(vmax, __shfl_xor(vmax, 16, 32));
    if (lane < 16) agg16[(size_t)m * 256 + n] = (_Float16)vmax;
  }
}

// ---- xt = agg @ Wr + br  ([4096,256]@[256,128]) ----------------------------
__global__ void xt_kernel(const _Float16* __restrict__ agg16,
                          const _Float16* __restrict__ Wrf, const float* __restrict__ br,
                          float* __restrict__ xt32, _Float16* __restrict__ xt16) {
  int rt = blockIdx.x;
  int tid = threadIdx.x, lane = tid & 31, w = tid >> 5;
  int row0 = rt * 16, half = lane >> 4;
  int ct = w;
  int n = ct * 16 + (lane & 15);
  v8f acc = {};
#pragma unroll
  for (int kt = 0; kt < 8; ++kt) {
    v16h a = a_frag(&agg16[(size_t)(row0 + (lane & 15)) * 256 + kt * 32], half);
    v16h b = ((const v16h*)Wrf)[(kt * 8 + ct) * 32 + lane];
    acc = wmma16(a, b, acc);
  }
  float bv = br[n];
#pragma unroll
  for (int r = 0; r < 8; ++r) {
    float v = acc[r] + bv;
    int row = row0 + r + 8 * half;
    xt32[(size_t)row * 128 + n] = v;
    xt16[(size_t)row * 128 + n] = (_Float16)v;
  }
}

// ---- qkv = xt @ in_proj_w + b  ([4096,128]@[128,384]) ----------------------
__global__ void qkv_kernel(const _Float16* __restrict__ xt16,
                           const _Float16* __restrict__ Wpf, const float* __restrict__ bp,
                           _Float16* __restrict__ qkv16) {
  int rt = blockIdx.x;
  int tid = threadIdx.x, lane = tid & 31, w = tid >> 5;
  int row0 = rt * 16, half = lane >> 4;
#pragma unroll
  for (int cc = 0; cc < 3; ++cc) {
    int ct = w + cc * 8;
    int n = ct * 16 + (lane & 15);
    v8f acc = {};
#pragma unroll
    for (int kt = 0; kt < 4; ++kt) {
      v16h a = a_frag(&xt16[(size_t)(row0 + (lane & 15)) * 128 + kt * 32], half);
      v16h b = ((const v16h*)Wpf)[(kt * 24 + ct) * 32 + lane];
      acc = wmma16(a, b, acc);
    }
    float bv = bp[n];
#pragma unroll
    for (int r = 0; r < 8; ++r)
      qkv16[(size_t)(row0 + r + 8 * half) * 384 + n] = (_Float16)(acc[r] + bv);
  }
}

// ---- swizzle K and V into B-fragment-major blocks (once, reused 256x) ------
// K: head h, 16-key tile kt: B[k=d][n=key]    -> block (h*256 + kt)
// V: head h, 32-key tile kt, d-tile t: B[k=key][n=d] -> block (h*256 + kt*2 + t)
__global__ void kv_swizzle_kernel(const _Float16* __restrict__ qkv16,
                                  _Float16* __restrict__ Kf,
                                  _Float16* __restrict__ Vf) {
  int lane = threadIdx.x, nn = lane & 15, half = lane >> 4;
  int b = blockIdx.x;
  if (b < 1024) {                     // K fragments
    int h = b >> 8, kt = b & 255;
    int key = kt * 16 + nn;
    const _Float16* krow = qkv16 + (size_t)key * 384 + 128 + h * 32;
    v16h v;
#pragma unroll
    for (int e = 0; e < 16; ++e) v[e] = krow[kofs(e, half)];
    ((v16h*)Kf)[(size_t)b * 32 + lane] = v;
  } else {                            // V fragments
    int b2 = b - 1024;
    int h = b2 >> 8, r = b2 & 255, kt = r >> 1, t = r & 1;
    v16h v;
#pragma unroll
    for (int e = 0; e < 16; ++e) {
      int key = kt * 32 + kofs(e, half);
      v[e] = qkv16[(size_t)key * 384 + 256 + h * 32 + t * 16 + nn];
    }
    ((v16h*)Vf)[(size_t)b2 * 32 + lane] = v;
  }
}

// ---- flash attention: one wave per (head, 16-query tile), Dh=32 ------------
__global__ void attn_kernel(const _Float16* __restrict__ qkv16,
                            const _Float16* __restrict__ Kf,
                            const _Float16* __restrict__ Vf,
                            _Float16* __restrict__ att16) {
  int b = blockIdx.x;
  int head = b & 3, qt = b >> 2;
  int lane = threadIdx.x, half = lane >> 4, nn = lane & 15;
  const float scale = 0.17677669529663687f; // 1/sqrt(32)
  __shared__ _Float16 P[16][32];
  v16h qf;
  {
    const _Float16* qrow = qkv16 + (size_t)(qt * 16 + nn) * 384 + head * 32;
#pragma unroll
    for (int e = 0; e < 16; ++e) qf[e] = qrow[kofs(e, half)];
  }
  float mrow[8], lrow[8];
  v8f o0 = {}, o1 = {};
#pragma unroll
  for (int r = 0; r < 8; ++r) { mrow[r] = -3.0e38f; lrow[r] = 0.f; }
  for (int it = 0; it < M_SMP / 32; ++it) {
    // S = Q K^T over 32 keys (two 16-key subtiles), fragments preswizzled
    v8f s0 = {}, s1 = {};
    {
      v16h kf0 = ((const v16h*)Kf)[(size_t)(head * 256 + it * 2 + 0) * 32 + lane];
      v16h kf1 = ((const v16h*)Kf)[(size_t)(head * 256 + it * 2 + 1) * 32 + lane];
      s0 = wmma16(qf, kf0, s0);
      s1 = wmma16(qf, kf1, s1);
    }
    // online softmax on C-fragment layout (row = r + 8*half, col = nn)
    float p0[8], p1[8];
#pragma unroll
    for (int r = 0; r < 8; ++r) {
      float a = s0[r] * scale, c = s1[r] * scale;
      float mx = fmaxf(a, c);
#pragma unroll
      for (int off = 8; off >= 1; off >>= 1) mx = fmaxf(mx, __shfl_xor(mx, off, 32));
      float mnew = fmaxf(mrow[r], mx);
      float alpha = __expf(mrow[r] - mnew);
      float e0 = __expf(a - mnew), e1 = __expf(c - mnew);
      float rs = e0 + e1;
#pragma unroll
      for (int off = 8; off >= 1; off >>= 1) rs += __shfl_xor(rs, off, 32);
      lrow[r] = lrow[r] * alpha + rs;
      mrow[r] = mnew;
      o0[r] *= alpha; o1[r] *= alpha;
      p0[r] = e0; p1[r] = e1;
    }
    __syncthreads();
#pragma unroll
    for (int r = 0; r < 8; ++r) {
      P[r + 8 * half][nn]      = (_Float16)p0[r];
      P[r + 8 * half][16 + nn] = (_Float16)p1[r];
    }
    __syncthreads();
    v16h pf = a_frag(&P[nn][0], half);
    {
      v16h vf0 = ((const v16h*)Vf)[(size_t)(head * 256 + it * 2 + 0) * 32 + lane];
      v16h vf1 = ((const v16h*)Vf)[(size_t)(head * 256 + it * 2 + 1) * 32 + lane];
      o0 = wmma16(pf, vf0, o0);
      o1 = wmma16(pf, vf1, o1);
    }
    __syncthreads();
  }
#pragma unroll
  for (int r = 0; r < 8; ++r) {
    float inv = 1.f / lrow[r];
    int row = qt * 16 + r + 8 * half;
    att16[(size_t)row * 128 + head * 32 + nn]      = (_Float16)(o0[r] * inv);
    att16[(size_t)row * 128 + head * 32 + 16 + nn] = (_Float16)(o1[r] * inv);
  }
}

// ---- out_proj + residual + layernorm1 --------------------------------------
__global__ void proj_ln1_kernel(const _Float16* __restrict__ att16,
                                const _Float16* __restrict__ Wof, const float* __restrict__ bo,
                                const float* __restrict__ xt32,
                                const float* __restrict__ g1, const float* __restrict__ be1,
                                float* __restrict__ y32, _Float16* __restrict__ y16) {
  int rt = blockIdx.x;
  int lane = threadIdx.x, half = lane >> 4, nn = lane & 15;
  int row0 = rt * 16;
  v8f acc[8];
#pragma unroll
  for (int ct = 0; ct < 8; ++ct) {
    v8f a = {};
#pragma unroll
    for (int kt = 0; kt < 4; ++kt) {
      v16h af = a_frag(&att16[(size_t)(row0 + nn) * 128 + kt * 32], half);
      v16h bf = ((const v16h*)Wof)[(kt * 8 + ct) * 32 + lane];
      a = wmma16(af, bf, a);
    }
    float bv = bo[ct * 16 + nn];
#pragma unroll
    for (int r = 0; r < 8; ++r)
      a[r] += bv + xt32[(size_t)(row0 + r + 8 * half) * 128 + ct * 16 + nn];
    acc[ct] = a;
  }
#pragma unroll
  for (int r = 0; r < 8; ++r) {
    float s = 0.f;
#pragma unroll
    for (int ct = 0; ct < 8; ++ct) s += acc[ct][r];
#pragma unroll
    for (int off = 8; off >= 1; off >>= 1) s += __shfl_xor(s, off, 32);
    float mu = s * (1.f / 128.f);
    float vs = 0.f;
#pragma unroll
    for (int ct = 0; ct < 8; ++ct) { float d = acc[ct][r] - mu; vs += d * d; }
#pragma unroll
    for (int off = 8; off >= 1; off >>= 1) vs += __shfl_xor(vs, off, 32);
    float rstd = rsqrtf(vs * (1.f / 128.f) + EPS_LN);
    int row = row0 + r + 8 * half;
#pragma unroll
    for (int ct = 0; ct < 8; ++ct) {
      int c = ct * 16 + nn;
      float v = (acc[ct][r] - mu) * rstd * g1[c] + be1[c];
      y32[(size_t)row * 128 + c] = v;
      y16[(size_t)row * 128 + c] = (_Float16)v;
    }
  }
}

// ---- MLP (128->256->128) + residual + layernorm2 -> final output -----------
__global__ void mlp_ln2_kernel(const _Float16* __restrict__ y16,
                               const float* __restrict__ y32,
                               const _Float16* __restrict__ Wm1f, const float* __restrict__ bm1,
                               const _Float16* __restrict__ Wm2f, const float* __restrict__ bm2,
                               const float* __restrict__ g2, const float* __restrict__ be2,
                               float* __restrict__ out) {
  int rt = blockIdx.x;
  int lane = threadIdx.x, half = lane >> 4, nn = lane & 15;
  int row0 = rt * 16;
  __shared__ _Float16 h1[16][256];
#pragma unroll
  for (int ct = 0; ct < 16; ++ct) {
    v8f a = {};
#pragma unroll
    for (int kt = 0; kt < 4; ++kt) {
      v16h af = a_frag(&y16[(size_t)(row0 + nn) * 128 + kt * 32], half);
      v16h bf = ((const v16h*)Wm1f)[(kt * 16 + ct) * 32 + lane];
      a = wmma16(af, bf, a);
    }
    float bv = bm1[ct * 16 + nn];
#pragma unroll
    for (int r = 0; r < 8; ++r) {
      float v = a[r] + bv;
      h1[r + 8 * half][ct * 16 + nn] = (_Float16)(v > 0.f ? v : 0.f);
    }
  }
  __syncthreads();
  v8f acc[8];
#pragma unroll
  for (int ct = 0; ct < 8; ++ct) {
    v8f a = {};
#pragma unroll
    for (int kt = 0; kt < 8; ++kt) {
      v16h af = a_frag(&h1[nn][kt * 32], half);
      v16h bf = ((const v16h*)Wm2f)[(kt * 8 + ct) * 32 + lane];
      a = wmma16(af, bf, a);
    }
    float bv = bm2[ct * 16 + nn];
#pragma unroll
    for (int r = 0; r < 8; ++r)
      a[r] += bv + y32[(size_t)(row0 + r + 8 * half) * 128 + ct * 16 + nn];
    acc[ct] = a;
  }
#pragma unroll
  for (int r = 0; r < 8; ++r) {
    float s = 0.f;
#pragma unroll
    for (int ct = 0; ct < 8; ++ct) s += acc[ct][r];
#pragma unroll
    for (int off = 8; off >= 1; off >>= 1) s += __shfl_xor(s, off, 32);
    float mu = s * (1.f / 128.f);
    float vs = 0.f;
#pragma unroll
    for (int ct = 0; ct < 8; ++ct) { float d = acc[ct][r] - mu; vs += d * d; }
#pragma unroll
    for (int off = 8; off >= 1; off >>= 1) vs += __shfl_xor(vs, off, 32);
    float rstd = rsqrtf(vs * (1.f / 128.f) + EPS_LN);
    int row = row0 + r + 8 * half;
#pragma unroll
    for (int ct = 0; ct < 8; ++ct) {
      int c = ct * 16 + nn;
      out[(size_t)row * 128 + c] = (acc[ct][r] - mu) * rstd * g2[c] + be2[c];
    }
  }
}

// ---- tail: q_pos and batch[idx] outputs ------------------------------------
__global__ void tail_kernel(const float* __restrict__ pos,
                            const int* __restrict__ idx,
                            const int* __restrict__ batch,
                            float* __restrict__ out) {
  int i = blockIdx.x * blockDim.x + threadIdx.x;
  if (i < M_SMP) {
    int q = idx[i];
    out[M_SMP * 128 + 3 * i + 0] = pos[3 * q + 0];
    out[M_SMP * 128 + 3 * i + 1] = pos[3 * q + 1];
    out[M_SMP * 128 + 3 * i + 2] = pos[3 * q + 2];
    out[M_SMP * 128 + 3 * M_SMP + i] = (float)batch[q];
  }
}

// ---- workspace layout (bytes, all 256B aligned) ----------------------------
#define OFF_COL    (size_t)0          // 4096*32*4            = 524288
#define OFF_W1F    (size_t)524288     // 24 blk * 1KB         = 24576
#define OFF_W2F    (size_t)548864     // 64 blk               = 65536
#define OFF_WRF    (size_t)614400     // 64 blk               = 65536
#define OFF_WPF    (size_t)679936     // 96 blk               = 98304
#define OFF_WOF    (size_t)778240     // 32 blk               = 32768
#define OFF_WM1F   (size_t)811008     // 64 blk               = 65536
#define OFF_WM2F   (size_t)876544     // 64 blk               = 65536
#define OFF_AGG16  (size_t)942080     // 4096*256*2           = 2097152
#define OFF_XT32   (size_t)3039232    // 4096*128*4           = 2097152
#define OFF_XT16   (size_t)5136384    // 4096*128*2           = 1048576
#define OFF_QKV16  (size_t)6184960    // 4096*384*2           = 3145728
#define OFF_ATT16  (size_t)9330688    // 4096*128*2           = 1048576
#define OFF_Y32    (size_t)10379264   // 4096*128*4           = 2097152
#define OFF_Y16    (size_t)12476416   // 4096*128*2           = 1048576
#define OFF_KF     (size_t)13524992   // 1024 blk * 1KB       = 1048576
#define OFF_VF     (size_t)14573568   // 1024 blk * 1KB       = 1048576

extern "C" void kernel_launch(void* const* d_in, const int* in_sizes, int n_in,
                              void* d_out, int out_size, void* d_ws, size_t ws_size,
                              hipStream_t stream) {
  const float* x    = (const float*)d_in[0];
  const float* pos  = (const float*)d_in[1];
  const int*   idx  = (const int*)d_in[2];
  const int*   bat  = (const int*)d_in[3];
  const float* W1   = (const float*)d_in[4];
  const float* b1   = (const float*)d_in[5];
  const float* W2   = (const float*)d_in[6];
  const float* b2   = (const float*)d_in[7];
  const float* Wr   = (const float*)d_in[8];
  const float* br   = (const float*)d_in[9];
  const float* Wp   = (const float*)d_in[10];
  const float* bp   = (const float*)d_in[11];
  const float* Wo   = (const float*)d_in[12];
  const float* bo   = (const float*)d_in[13];
  const float* Wm1  = (const float*)d_in[14];
  const float* bm1  = (const float*)d_in[15];
  const float* Wm2  = (const float*)d_in[16];
  const float* bm2  = (const float*)d_in[17];
  const float* g1   = (const float*)d_in[18];
  const float* be1  = (const float*)d_in[19];
  const float* g2   = (const float*)d_in[20];
  const float* be2  = (const float*)d_in[21];

  char* ws = (char*)d_ws;
  int*       colW  = (int*)(ws + OFF_COL);
  _Float16*  W1f   = (_Float16*)(ws + OFF_W1F);
  _Float16*  W2f   = (_Float16*)(ws + OFF_W2F);
  _Float16*  Wrf   = (_Float16*)(ws + OFF_WRF);
  _Float16*  Wpf   = (_Float16*)(ws + OFF_WPF);
  _Float16*  Wof   = (_Float16*)(ws + OFF_WOF);
  _Float16*  Wm1f  = (_Float16*)(ws + OFF_WM1F);
  _Float16*  Wm2f  = (_Float16*)(ws + OFF_WM2F);
  _Float16*  agg16 = (_Float16*)(ws + OFF_AGG16);
  float*     xt32  = (float*)(ws + OFF_XT32);
  _Float16*  xt16  = (_Float16*)(ws + OFF_XT16);
  _Float16*  qkv16 = (_Float16*)(ws + OFF_QKV16);
  _Float16*  att16 = (_Float16*)(ws + OFF_ATT16);
  float*     y32   = (float*)(ws + OFF_Y32);
  _Float16*  y16   = (_Float16*)(ws + OFF_Y16);
  _Float16*  Kf    = (_Float16*)(ws + OFF_KF);
  _Float16*  Vf    = (_Float16*)(ws + OFF_VF);
  float*     out   = (float*)d_out;

  // weight swizzles (B-fragment-major, K padded to multiple of 32)
  frag_convert_kernel<<<3 * 8,  32, 0, stream>>>(W1,  W1f,  67,  128); // K 67->96
  frag_convert_kernel<<<4 * 16, 32, 0, stream>>>(W2,  W2f,  128, 256);
  frag_convert_kernel<<<8 * 8,  32, 0, stream>>>(Wr,  Wrf,  256, 128);
  frag_convert_kernel<<<4 * 24, 32, 0, stream>>>(Wp,  Wpf,  128, 384);
  frag_convert_kernel<<<4 * 8,  32, 0, stream>>>(Wo,  Wof,  128, 128);
  frag_convert_kernel<<<4 * 16, 32, 0, stream>>>(Wm1, Wm1f, 128, 256);
  frag_convert_kernel<<<8 * 8,  32, 0, stream>>>(Wm2, Wm2f, 256, 128);

  knn_kernel<<<M_SMP, 32, 0, stream>>>(pos, idx, colW);
  conv_kernel<<<M_SMP, 256, 0, stream>>>(x, pos, idx, colW, W1f, b1, W2f, b2, agg16);
  xt_kernel<<<M_SMP / 16, 256, 0, stream>>>(agg16, Wrf, br, xt32, xt16);
  qkv_kernel<<<M_SMP / 16, 256, 0, stream>>>(xt16, Wpf, bp, qkv16);
  kv_swizzle_kernel<<<2048, 32, 0, stream>>>(qkv16, Kf, Vf);
  attn_kernel<<<4 * (M_SMP / 16), 32, 0, stream>>>(qkv16, Kf, Vf, att16);
  proj_ln1_kernel<<<M_SMP / 16, 32, 0, stream>>>(att16, Wof, bo, xt32, g1, be1, y32, y16);
  mlp_ln2_kernel<<<M_SMP / 16, 32, 0, stream>>>(y16, y32, Wm1f, bm1, Wm2f, bm2, g2, be2, out);
  tail_kernel<<<(M_SMP + 255) / 256, 256, 0, stream>>>(pos, idx, bat, out);
}
